// CIN2_52725018526224
// MI455X (gfx1250) — matrix operand order repaired and checked by reading, not compiled, not run on hardware
//
#include <hip/hip_runtime.h>

typedef _Float16 v16h __attribute__((ext_vector_type(16)));
typedef _Float16 v8h  __attribute__((ext_vector_type(8)));
typedef float    v8f  __attribute__((ext_vector_type(8)));
typedef float    f32x2 __attribute__((ext_vector_type(2)));

#define B_TOT    2048
#define F0       39
#define S        200
#define OPAD     208      // 13 real tiles of 16 rows (rows 200..207 zero-padded)
#define NWAVES   4
#define TPW      4        // o-tiles per wave: 16 logical tiles, >=OPAD clamped
#define NTHREADS 128

#define FANIN0   1521     // 39*39
#define FANPAD0  1536
#define FANIN12  3900     // 100*39
#define FANPAD12 3904

#define T0   (OPAD * FANPAD0)    // 319488 elements
#define T12  (OPAD * FANPAD12)   // 812032 elements
#define OFF0 0
#define OFF1 T0
#define OFF2 (T0 + T12)
#define WTOT (T0 + 2 * T12)      // elements per (hi|lo) plane

// ---------------------------------------------------------------------------
// Prep: W[c,o] (f32) -> Wt[o,c] padded to [OPAD][FANPAD], split into f16 hi/lo
// ---------------------------------------------------------------------------
__global__ void __launch_bounds__(256)
prep_w(const float* __restrict__ W0, const float* __restrict__ W1,
       const float* __restrict__ W2, _Float16* __restrict__ Whi,
       _Float16* __restrict__ Wlo)
{
    int idx = blockIdx.x * blockDim.x + threadIdx.x;
    if (idx >= WTOT) return;
    float w = 0.0f;
    if (idx < T0) {
        int o = idx / FANPAD0, c = idx % FANPAD0;
        if (o < S && c < FANIN0) w = W0[c * S + o];
    } else if (idx < OFF2) {
        int r = idx - OFF1;
        int o = r / FANPAD12, c = r % FANPAD12;
        if (o < S && c < FANIN12) w = W1[c * S + o];
    } else {
        int r = idx - OFF2;
        int o = r / FANPAD12, c = r % FANPAD12;
        if (o < S && c < FANIN12) w = W2[c * S + o];
    }
    _Float16 hi = (_Float16)w;
    _Float16 lo = (_Float16)(w - (float)hi);
    Whi[idx] = hi;
    Wlo[idx] = lo;
}

// ---------------------------------------------------------------------------
// One K=32 chunk: build B fragments once (Z = x*h, hi/lo split), then run
// 6 WMMAs on each of the wave's TPW o-tiles (A loaded per tile).
// ---------------------------------------------------------------------------
template<int HN, int FANIN, int FANPAD, bool CLAMP>
__device__ __forceinline__ void do_chunk(
    int cb, int half, int row,
    const _Float16* __restrict__ Whi, const _Float16* __restrict__ Wlo,
    int wave, const f32x2* xpk, const f32x2* hpk,
    v8f (&acc)[TPW][2])
{
    // ---- B fragments: lanes 0-15 hold K=0..15, lanes 16-31 hold K=16..31
    const int bbase = cb + half * 16;
    v16h q0h, q0l, q1h, q1l;
    if constexpr (!CLAMP) {
        // c = bbase+e crosses at most one m-boundary (HN > 16): one divmod,
        // two candidate x rows, per-element select.
        const int m0    = bbase / HN;
        const int n0    = bbase - m0 * HN;
        const int split = HN - n0;               // e < split -> m0, else m0+1
        const f32x2 xv0 = xpk[m0 * 16 + row];
        const f32x2 xv1 = xpk[(m0 + 1) * 16 + row];
#pragma unroll
        for (int e = 0; e < 16; ++e) {
            const bool wrap = (e >= split);
            const int  n    = wrap ? (e - split) : (n0 + e);
            const f32x2 xv  = wrap ? xv1 : xv0;
            const f32x2 hv  = hpk[n * 16 + row];
            float p0 = xv[0] * hv[0];
            float p1 = xv[1] * hv[1];
            _Float16 h0 = (_Float16)p0, h1 = (_Float16)p1;
            q0h[e] = h0; q0l[e] = (_Float16)(p0 - (float)h0);
            q1h[e] = h1; q1l[e] = (_Float16)(p1 - (float)h1);
        }
    } else {
#pragma unroll
        for (int e = 0; e < 16; ++e) {
            int c = bbase + e;
            c = (c < FANIN) ? c : 0;             // A rows are zero there
            const int m = c / HN;
            const int n = c - m * HN;
            const f32x2 xv = xpk[m * 16 + row];
            const f32x2 hv = hpk[n * 16 + row];
            float p0 = xv[0] * hv[0];
            float p1 = xv[1] * hv[1];
            _Float16 h0 = (_Float16)p0, h1 = (_Float16)p1;
            q0h[e] = h0; q0l[e] = (_Float16)(p0 - (float)h0);
            q1h[e] = h1; q1l[e] = (_Float16)(p1 - (float)h1);
        }
    }

    // ---- A fragments (half 0 -> K 0..15, half 1 -> K 16..31)
    const int abase = cb + half * 16;
#pragma unroll
    for (int t = 0; t < TPW; ++t) {
        int orow = (wave + NWAVES * t) * 16 + row;
        orow = (orow < OPAD) ? orow : (OPAD - 1);   // pad tiles: results discarded
        const size_t roff = (size_t)orow * FANPAD;
        v8h ph0 = *(const v8h*)(Whi + roff + abase);
        v8h ph1 = *(const v8h*)(Whi + roff + abase + 8);
        v8h pl0 = *(const v8h*)(Wlo + roff + abase);
        v8h pl1 = *(const v8h*)(Wlo + roff + abase + 8);
        v16h ahi, alo;
#pragma unroll
        for (int j = 0; j < 8; ++j) {
            ahi[j] = ph0[j]; ahi[8 + j] = ph1[j];
            alo[j] = pl0[j]; alo[8 + j] = pl1[j];
        }
        acc[t][0] = __builtin_amdgcn_wmma_f32_16x16x32_f16(false, ahi, false, q0h, (short)0, acc[t][0], false, false);
        acc[t][0] = __builtin_amdgcn_wmma_f32_16x16x32_f16(false, alo, false, q0h, (short)0, acc[t][0], false, false);
        acc[t][0] = __builtin_amdgcn_wmma_f32_16x16x32_f16(false, ahi, false, q0l, (short)0, acc[t][0], false, false);
        acc[t][1] = __builtin_amdgcn_wmma_f32_16x16x32_f16(false, ahi, false, q1h, (short)0, acc[t][1], false, false);
        acc[t][1] = __builtin_amdgcn_wmma_f32_16x16x32_f16(false, alo, false, q1h, (short)0, acc[t][1], false, false);
        acc[t][1] = __builtin_amdgcn_wmma_f32_16x16x32_f16(false, ahi, false, q1l, (short)0, acc[t][1], false, false);
    }
}

// ---------------------------------------------------------------------------
// One CIN layer for one batch element (4 waves x 4 o-tiles)
// ---------------------------------------------------------------------------
template<int HN, int FANIN, int FANPAD, bool WRITE_HIDDEN>
__device__ __forceinline__ void cin_layer(
    const _Float16* __restrict__ Whi, const _Float16* __restrict__ Wlo,
    const float* __restrict__ bias,
    const f32x2* xpk, const f32x2* hpk, f32x2* hout,
    float* cur, float* __restrict__ outp,
    int dstart, int dcount)
{
    const int tid  = threadIdx.x;
    const int lane = tid & 31;
    const int wave = tid >> 5;          // 0..3
    const int row  = lane & 15;
    const int half = lane >> 4;

    v8f acc[TPW][2];
#pragma unroll
    for (int t = 0; t < TPW; ++t) {
        v8f z = {};
        acc[t][0] = z;
        acc[t][1] = z;
    }

    constexpr int KMAIN = (FANIN / 32) * 32;
    for (int cb = 0; cb < KMAIN; cb += 32)
        do_chunk<HN, FANIN, FANPAD, false>(cb, half, row, Whi, Wlo, wave, xpk, hpk, acc);
    for (int cb = KMAIN; cb < FANPAD; cb += 32)
        do_chunk<HN, FANIN, FANPAD, true>(cb, half, row, Whi, Wlo, wave, xpk, hpk, acc);

    // Epilogue: bias + ReLU; C/D layout (M = r + 8*half, N = row) -> LDS
#pragma unroll
    for (int t = 0; t < TPW; ++t) {
        const int obase = (wave + NWAVES * t) * 16;
#pragma unroll
        for (int r = 0; r < 8; ++r) {
            const int o = obase + r + 8 * half;
            if (o < S) {
                float bv = bias[o];
                float v0 = acc[t][0][r] + bv; v0 = v0 > 0.0f ? v0 : 0.0f;
                float v1 = acc[t][1][r] + bv; v1 = v1 > 0.0f ? v1 : 0.0f;
                cur[o * 32 + row]      = v0;
                cur[o * 32 + 16 + row] = v1;
            }
        }
    }
    __syncthreads();

    if constexpr (WRITE_HIDDEN) {
        for (int i = tid; i < 100 * 16; i += NTHREADS) {
            int o = i >> 4, dd = i & 15;
            f32x2 v;
            v[0] = cur[o * 32 + dd];
            v[1] = cur[o * 32 + 16 + dd];
            hout[i] = v;
        }
    }
    for (int i = tid; i < dcount; i += NTHREADS) {
        const float* cr = cur + (dstart + i) * 32;
        float s = 0.0f;
#pragma unroll
        for (int d = 0; d < 32; ++d) s += cr[d];
        outp[i] = s;
    }
    __syncthreads();
}

// ---------------------------------------------------------------------------
// Main kernel: one block per batch element, 3 fused layers
// ---------------------------------------------------------------------------
__global__ void __launch_bounds__(NTHREADS)
cin_main(const float* __restrict__ x,
         const float* __restrict__ b0, const float* __restrict__ b1,
         const float* __restrict__ b2,
         const _Float16* __restrict__ Whi, const _Float16* __restrict__ Wlo,
         float* __restrict__ out)
{
    __shared__ f32x2 xpk[F0 * 16];    // x[b] packed as (d, d+16) pairs
    __shared__ f32x2 hpk[100 * 16];   // hidden packed the same way
    __shared__ float cur[OPAD * 32];

    const int b   = blockIdx.x;
    const int tid = threadIdx.x;
    const float* xb = x + (size_t)b * (F0 * 32);
    for (int i = tid; i < F0 * 16; i += NTHREADS) {
        int m = i >> 4, dd = i & 15;
        f32x2 v;
        v[0] = xb[m * 32 + dd];
        v[1] = xb[m * 32 + 16 + dd];
        xpk[i] = v;
    }
    __syncthreads();

    float* outb = out + (size_t)b * 400;
    // layer 0: h = x, direct = cur[100:200] -> out[0:100], hidden = cur[0:100]
    cin_layer<39,  FANIN0,  FANPAD0,  true >(Whi + OFF0, Wlo + OFF0, b0, xpk, xpk, hpk, cur, outb,       100, 100);
    // layer 1: direct = cur[100:200] -> out[100:200], hidden = cur[0:100]
    cin_layer<100, FANIN12, FANPAD12, true >(Whi + OFF1, Wlo + OFF1, b1, xpk, hpk, hpk, cur, outb + 100, 100, 100);
    // layer 2: direct = cur[0:200] -> out[200:400]
    cin_layer<100, FANIN12, FANPAD12, false>(Whi + OFF2, Wlo + OFF2, b2, xpk, hpk, hpk, cur, outb + 200, 0,   200);
}

// ---------------------------------------------------------------------------
extern "C" void kernel_launch(void* const* d_in, const int* in_sizes, int n_in,
                              void* d_out, int out_size, void* d_ws, size_t ws_size,
                              hipStream_t stream)
{
    (void)in_sizes; (void)n_in; (void)out_size;
    const float* x  = (const float*)d_in[0];
    const float* W0 = (const float*)d_in[1];
    const float* b0 = (const float*)d_in[2];
    const float* W1 = (const float*)d_in[3];
    const float* b1 = (const float*)d_in[4];
    const float* W2 = (const float*)d_in[5];
    const float* b2 = (const float*)d_in[6];
    float* out = (float*)d_out;

    if (ws_size < (size_t)2 * WTOT * sizeof(_Float16)) return;
    _Float16* Whi = (_Float16*)d_ws;
    _Float16* Wlo = Whi + WTOT;

    int pgrid = (WTOT + 255) / 256;
    prep_w<<<pgrid, 256, 0, stream>>>(W0, W1, W2, Whi, Wlo);
    cin_main<<<B_TOT, NTHREADS, 0, stream>>>(x, b0, b1, b2, Whi, Wlo, out);
}
